// MambaController_69243462746250
// MI455X (gfx1250) — matrix operand order repaired and checked by reading, not compiled, not run on hardware
//
#include <hip/hip_runtime.h>

#define B_   4
#define T_   2048
#define V_   32000
#define DM_  768
#define L_   4
#define DI_  1536
#define BT_  (B_*T_)

typedef __bf16 bf16_t;
typedef __attribute__((ext_vector_type(16))) __bf16 v16bf;
typedef __attribute__((ext_vector_type(8)))  __bf16 v8bf;
typedef __attribute__((ext_vector_type(4)))  __bf16 v4bf;
typedef __attribute__((ext_vector_type(8)))  float  v8f;

// --------------------------------------------------------------- utilities
// W (L batched, K x N) f32 -> WT (L batched, N x K) bf16, tiled transpose.
__global__ __launch_bounds__(256) void transpose_to_bf16_kernel(
    const float* __restrict__ s, bf16_t* __restrict__ d, int K, int N) {
  __shared__ float tile[32][33];
  const int l  = blockIdx.z;
  const int kb = blockIdx.y * 32;
  const int nb = blockIdx.x * 32;
  const float* sl = s + (size_t)l * K * N;
  bf16_t*      dl = d + (size_t)l * N * K;
  const int tx = threadIdx.x & 31, ty = threadIdx.x >> 5;   // 32 x 8
#pragma unroll
  for (int i = ty; i < 32; i += 8)
    tile[i][tx] = sl[(size_t)(kb + i) * N + nb + tx];
  __syncthreads();
#pragma unroll
  for (int i = ty; i < 32; i += 8)
    dl[(size_t)(nb + i) * K + kb + tx] = (bf16_t)tile[tx][i];
}

__global__ void embed_kernel(const int* __restrict__ tokens,
                             const float* __restrict__ ew,
                             float* __restrict__ h) {
  const int row = blockIdx.x;                 // 0..BT-1
  const int tok = tokens[row];
  const float* src = ew + (size_t)tok * DM_;
  float* dst = h + (size_t)row * DM_;
  for (int i = threadIdx.x; i < DM_; i += blockDim.x) dst[i] = src[i];
}

// ------------------------------------------------------- WMMA GEMM (bf16)
// C[M,N] = A[M,K] * B[K,N] + bias[N] (+ res[M,N]); B given TRANSPOSED (NxK).
// Block tile 128x128, 8 waves; each wave: 32x64 = 2x4 WMMA 16x16 tiles.
// Double-buffered LDS; next tile's global loads issued before the WMMAs.
template <bool A_IS_F32, bool RES>
__global__ __launch_bounds__(256) void gemm_wmma_kernel(
    const void* __restrict__ Av, const bf16_t* __restrict__ BwT,
    const float* __restrict__ bias, const float* res,
    float* C, int M, int N, int K) {
  __shared__ bf16_t As[2][128 * 32];   // [m][k]
  __shared__ bf16_t Bs[2][128 * 32];   // [n][k]

  const int row0 = blockIdx.y * 128;
  const int col0 = blockIdx.x * 128;
  const int tid  = threadIdx.x;
  const int lane = tid & 31;
  const int wave = tid >> 5;
  const int wm   = wave & 3;           // 4 waves along M (32 rows each)
  const int wn   = wave >> 2;          // 2 waves along N (64 cols each)
  const int hsel = (lane >> 4) & 1;    // lane-half selects K-subset
  const int l16  = lane & 15;

  const float*  Af = (const float*)Av;
  const bf16_t* Ab = (const bf16_t*)Av;

  v8f acc[2][4];
#pragma unroll
  for (int i = 0; i < 2; ++i)
#pragma unroll
    for (int j = 0; j < 4; ++j)
#pragma unroll
      for (int r = 0; r < 8; ++r) acc[i][j][r] = 0.f;

  // staging decomposition: 8 k-groups of 4 elements x 32 row-slots x 4 steps
  const int a_kg = tid & 7;
  const int a_r0 = tid >> 3;

  float4 areg[4];
  v4bf   aregb[4];
  v4bf   breg[4];

  auto load_tile = [&](int kt) {
    const int k0 = kt << 5;
#pragma unroll
    for (int rr = 0; rr < 4; ++rr) {
      const int r = a_r0 + rr * 32;
      if (A_IS_F32)
        areg[rr] = *(const float4*)(Af + (size_t)(row0 + r) * K + k0 + a_kg * 4);
      else
        aregb[rr] = *(const v4bf*)(Ab + (size_t)(row0 + r) * K + k0 + a_kg * 4);
      breg[rr] = *(const v4bf*)(BwT + (size_t)(col0 + r) * K + k0 + a_kg * 4);
    }
  };
  auto store_tile = [&](int buf) {
#pragma unroll
    for (int rr = 0; rr < 4; ++rr) {
      const int r = a_r0 + rr * 32;
      if (A_IS_F32) {
        v4bf v;
        v[0] = (bf16_t)areg[rr].x; v[1] = (bf16_t)areg[rr].y;
        v[2] = (bf16_t)areg[rr].z; v[3] = (bf16_t)areg[rr].w;
        *(v4bf*)&As[buf][r * 32 + a_kg * 4] = v;
      } else {
        *(v4bf*)&As[buf][r * 32 + a_kg * 4] = aregb[rr];
      }
      *(v4bf*)&Bs[buf][r * 32 + a_kg * 4] = breg[rr];
    }
  };

  const int nkt = K >> 5;
  load_tile(0);
  store_tile(0);
  __syncthreads();

  for (int kt = 0; kt < nkt; ++kt) {
    const int buf = kt & 1;
    if (kt + 1 < nkt) load_tile(kt + 1);           // loads in flight
    if (kt + 2 < nkt)                               // global_prefetch_b8
      __builtin_prefetch(BwT + (size_t)(col0 + a_r0) * K + ((kt + 2) << 5), 0, 1);

    // ---- fragments per ISA 7.12.2 layouts
    v16bf afr[2], bfr[4];
#pragma unroll
    for (int mf = 0; mf < 2; ++mf) {
      const bf16_t* ap = &As[buf][(wm * 32 + mf * 16 + l16) * 32];
      v8bf lo = *(const v8bf*)(ap + 8 * hsel);        // K = 8g..8g+7
      v8bf hi = *(const v8bf*)(ap + 16 + 8 * hsel);   // K = 16+8g..
      afr[mf] = __builtin_shufflevector(lo, hi, 0, 1, 2, 3, 4, 5, 6, 7,
                                        8, 9, 10, 11, 12, 13, 14, 15);
    }
#pragma unroll
    for (int nf = 0; nf < 4; ++nf) {
      const bf16_t* bp = &Bs[buf][(wn * 64 + nf * 16 + l16) * 32];
      v8bf lo = *(const v8bf*)(bp + 8 * hsel);        // K = 8g..8g+7
      v8bf hi = *(const v8bf*)(bp + 16 + 8 * hsel);   // K = 16+8g..
      bfr[nf] = __builtin_shufflevector(lo, hi, 0, 1, 2, 3, 4, 5, 6, 7,
                                        8, 9, 10, 11, 12, 13, 14, 15);
    }
#pragma unroll
    for (int mf = 0; mf < 2; ++mf)
#pragma unroll
      for (int nf = 0; nf < 4; ++nf)
        acc[mf][nf] = __builtin_amdgcn_wmma_f32_16x16x32_bf16(
            false, afr[mf], false, bfr[nf], (short)0, acc[mf][nf],
            false, false);

    if (kt + 1 < nkt) store_tile(buf ^ 1);           // cvt + ds_store
    __syncthreads();
  }

  // ---- epilogue: C/D layout row = vgpr + 8*(lane/16), col = lane%16
#pragma unroll
  for (int mf = 0; mf < 2; ++mf) {
#pragma unroll
    for (int nf = 0; nf < 4; ++nf) {
      const int col = col0 + wn * 64 + nf * 16 + l16;
      const float bv = bias[col];
#pragma unroll
      for (int r = 0; r < 8; ++r) {
        const int row = row0 + wm * 32 + mf * 16 + hsel * 8 + r;
        const size_t idx = (size_t)row * N + col;
        float v = acc[mf][nf][r] + bv;
        if (RES) v += res[idx];
        C[idx] = v;
      }
    }
  }
}

// ------------------------------------- fused causal conv + SiLU + EMA + gate
__global__ __launch_bounds__(256) void scan_kernel(
    const float* __restrict__ xz,   // [BT, 2*DI] : x_inner | z
    const float* __restrict__ cw,   // [DI, 4]
    const float* __restrict__ cb,   // [DI]
    const float* __restrict__ Dp,   // [DI]
    bf16_t* __restrict__ y) {       // [BT, DI]
  const int idx = blockIdx.x * blockDim.x + threadIdx.x;
  if (idx >= B_ * DI_) return;
  const int b = idx / DI_, ch = idx % DI_;
  const float w0 = cw[ch * 4 + 0], w1 = cw[ch * 4 + 1];
  const float w2 = cw[ch * 4 + 2], w3 = cw[ch * 4 + 3];
  const float bb = cb[ch], dp = Dp[ch];
  const float* xr = xz + (size_t)b * T_ * (2 * DI_) + ch;
  const float* zr = xr + DI_;
  bf16_t* yr = y + (size_t)b * T_ * DI_ + ch;
  float x0 = 0.f, x1 = 0.f, x2 = 0.f, hs = 0.f;
  for (int t0 = 0; t0 < T_; t0 += 16) {
    float xa[16], za[16];
#pragma unroll
    for (int i = 0; i < 16; ++i) {
      const size_t o = (size_t)(t0 + i) * (2 * DI_);
      xa[i] = xr[o];
      za[i] = zr[o];
    }
#pragma unroll
    for (int i = 0; i < 16; ++i) {
      const float xt = xa[i];
      const float c  = w0 * x0 + w1 * x1 + w2 * x2 + w3 * xt + bb;
      const float xc = c / (1.f + __expf(-c));        // SiLU
      hs = 0.9f * hs + 0.1f * xc;                     // EMA (post-update)
      const float z  = za[i];
      const float sz = z / (1.f + __expf(-z));
      yr[(size_t)(t0 + i) * DI_] = (bf16_t)((xc * dp + hs) * sz);
      x0 = x1; x1 = x2; x2 = xt;
    }
  }
}

// ------------------------------------------------------------- LayerNorm
__global__ __launch_bounds__(256) void ln_kernel(
    const float* __restrict__ h, const float* __restrict__ g,
    const float* __restrict__ b, bf16_t* __restrict__ hn) {
  const int row = blockIdx.x;
  const float* x = h + (size_t)row * DM_;
  float s = 0.f, s2 = 0.f;
  for (int i = threadIdx.x; i < DM_; i += 256) {
    const float v = x[i];
    s += v; s2 += v * v;
  }
#pragma unroll
  for (int off = 16; off > 0; off >>= 1) {
    s  += __shfl_xor(s, off, 32);
    s2 += __shfl_xor(s2, off, 32);
  }
  __shared__ float ps[8], ps2[8];
  const int wv = threadIdx.x >> 5;
  if ((threadIdx.x & 31) == 0) { ps[wv] = s; ps2[wv] = s2; }
  __syncthreads();
  float ts = 0.f, ts2 = 0.f;
#pragma unroll
  for (int i = 0; i < 8; ++i) { ts += ps[i]; ts2 += ps2[i]; }
  const float mu  = ts * (1.f / DM_);
  const float var = ts2 * (1.f / DM_) - mu * mu;
  const float inv = rsqrtf(var + 1e-5f);
  bf16_t* o = hn + (size_t)row * DM_;
  for (int i = threadIdx.x; i < DM_; i += 256)
    o[i] = (bf16_t)((x[i] - mu) * inv * g[i] + b[i]);
}

// ---------------------------------------------------------------- launcher
extern "C" void kernel_launch(void* const* d_in, const int* in_sizes, int n_in,
                              void* d_out, int out_size, void* d_ws,
                              size_t ws_size, hipStream_t stream) {
  (void)in_sizes; (void)n_in; (void)out_size; (void)ws_size;
  const int*   tokens  = (const int*)d_in[0];
  const float* embed_w = (const float*)d_in[1];
  const float* in_w    = (const float*)d_in[2];
  const float* in_b    = (const float*)d_in[3];
  const float* conv_w  = (const float*)d_in[4];
  const float* conv_b  = (const float*)d_in[5];
  const float* D_p     = (const float*)d_in[6];
  const float* out_w   = (const float*)d_in[7];
  const float* out_b   = (const float*)d_in[8];
  const float* ln_g    = (const float*)d_in[9];
  const float* ln_b    = (const float*)d_in[10];
  const float* head_w  = (const float*)d_in[11];
  const float* head_b  = (const float*)d_in[12];

  char* ws = (char*)d_ws;
  size_t off = 0;
  auto alloc = [&](size_t bytes) -> char* {
    char* p = ws + off;
    off = (off + bytes + 255) & ~(size_t)255;
    return p;
  };
  float*  h      = (float*)alloc((size_t)BT_ * DM_ * 4);
  float*  xz     = (float*)alloc((size_t)BT_ * 2 * DI_ * 4);
  bf16_t* yb     = (bf16_t*)alloc((size_t)BT_ * DI_ * 2);
  bf16_t* hn     = (bf16_t*)alloc((size_t)BT_ * DM_ * 2);
  bf16_t* winT   = (bf16_t*)alloc((size_t)L_ * DM_ * 2 * DI_ * 2);  // (L,2DI,DM)
  bf16_t* woutT  = (bf16_t*)alloc((size_t)L_ * DI_ * DM_ * 2);      // (L,DM,DI)
  bf16_t* whT    = (bf16_t*)alloc((size_t)DM_ * V_ * 2);            // (V,DM)

  // weights -> bf16, transposed to (N,K), once per launch
  transpose_to_bf16_kernel<<<dim3(2 * DI_ / 32, DM_ / 32, L_), 256, 0, stream>>>(
      in_w, winT, DM_, 2 * DI_);
  transpose_to_bf16_kernel<<<dim3(DM_ / 32, DI_ / 32, L_), 256, 0, stream>>>(
      out_w, woutT, DI_, DM_);
  transpose_to_bf16_kernel<<<dim3(V_ / 32, DM_ / 32, 1), 256, 0, stream>>>(
      head_w, whT, DM_, V_);

  embed_kernel<<<BT_, 256, 0, stream>>>(tokens, embed_w, h);

  for (int l = 0; l < L_; ++l) {
    // xz = h @ in_w[l] + in_b[l]     (M=8192, N=3072, K=768)
    gemm_wmma_kernel<true, false>
        <<<dim3(2 * DI_ / 128, BT_ / 128), 256, 0, stream>>>(
            h, winT + (size_t)l * DM_ * 2 * DI_, in_b + (size_t)l * 2 * DI_,
            nullptr, xz, BT_, 2 * DI_, DM_);
    // fused conv/SiLU/EMA/gate -> y (bf16)
    scan_kernel<<<(B_ * DI_ + 255) / 256, 256, 0, stream>>>(
        xz, conv_w + (size_t)l * DI_ * 4, conv_b + (size_t)l * DI_,
        D_p + (size_t)l * DI_, yb);
    // h += y @ out_w[l] + out_b[l]   (M=8192, N=768, K=1536)
    gemm_wmma_kernel<false, true>
        <<<dim3(DM_ / 128, BT_ / 128), 256, 0, stream>>>(
            yb, woutT + (size_t)l * DI_ * DM_, out_b + (size_t)l * DM_, h, h,
            BT_, DM_, DI_);
  }

  ln_kernel<<<BT_, 256, 0, stream>>>(h, ln_g, ln_b, hn);

  // logits = hn @ head_w + head_b    (M=8192, N=32000, K=768)
  gemm_wmma_kernel<false, false>
      <<<dim3(V_ / 128, BT_ / 128), 256, 0, stream>>>(
          hn, whT, head_b, nullptr, (float*)d_out, BT_, V_, DM_);
}
